// SimplifiedRoutedExperts_46420006535174
// MI455X (gfx1250) — matrix operand order repaired and checked by reading, not compiled, not run on hardware
//
#include <hip/hip_runtime.h>
#include <hip/hip_bf16.h>

// Problem sizes (fixed by the reference)
#define B_  2
#define E_  16
#define S_  512
#define H_  2048
#define I_  1408
#define MPE (B_ * S_)   // 1024 rows (tokens) per expert

typedef __attribute__((ext_vector_type(16))) __bf16 v16bf;
typedef __attribute__((ext_vector_type(8)))  float  v8f;

union FragBF { v16bf v; uint4 q[2]; };

__device__ __forceinline__ unsigned f2u(float f) {
  union { float f; unsigned u; } c; c.f = f; return c.u;
}
// Pack two floats to two bf16 (round-half-up) in one v_perm_b32:
// result = { bf16(hi), bf16(lo) }
__device__ __forceinline__ unsigned packbf(float hi, float lo) {
  return __builtin_amdgcn_perm(f2u(hi) + 0x8000u, f2u(lo) + 0x8000u, 0x07060302u);
}
__device__ __forceinline__ uint2 f2bf4(float4 v) {
  uint2 r; r.x = packbf(v.y, v.x); r.y = packbf(v.w, v.z); return r;
}
__device__ __forceinline__ unsigned short f2bf(float f) {
  return (unsigned short)((f2u(f) + 0x8000u) >> 16);
}

// Async global -> LDS copy, 16B per lane (ASYNCcnt-tracked).
// lds_addr: LDS byte address (low 32 bits of generic shared pointer).
__device__ __forceinline__ void async_copy_b128(unsigned lds_addr, const void* g) {
  asm volatile("global_load_async_to_lds_b128 %0, %1, off"
               :: "v"(lds_addr), "v"(g) : "memory");
}
#define WAIT_ASYNCCNT(n) asm volatile("s_wait_asynccnt %0" :: "n"(n) : "memory")

// LDS tiles are [128 rows][40 ushorts] (32 data + 8 pad -> 80B pitch, conflict-free b128 reads)
#define LPITCH 40

// A-fragment (16x32 bf16, ISA 7.12.2): lane<16 holds K{0..7,16..23} of row M,
// lane>=16 holds K{8..15,24..31}.
__device__ __forceinline__ v16bf frag_a(const unsigned short* tile, int row, int lane) {
  const unsigned short* p = tile + row * LPITCH + ((lane & 16) ? 8 : 0);
  FragBF f;
  f.q[0] = *(const uint4*)(p);
  f.q[1] = *(const uint4*)(p + 16);
  return f.v;
}
// B-fragment (32x16 bf16): lane<16 holds K{0..7,16..23} of col N=lane%16, lane>=16 holds K{8..15,24..31}.
// Tile stored as [N][K] row-major.
__device__ __forceinline__ v16bf frag_b(const unsigned short* tile, int col, int lane) {
  const unsigned short* p = tile + col * LPITCH + ((lane & 16) ? 8 : 0);
  FragBF f;
  f.q[0] = *(const uint4*)(p);
  f.q[1] = *(const uint4*)(p + 16);
  return f.v;
}

// ---------------------------------------------------------------------------
// Kernel 1: act[e,m,i] = silu(x @ w1^T) * (x @ w3^T)   (bf16 output to ws)
// Tile: 128(M) x 128(N=I), K=H stepped by 32. 8 waves in 2x4 grid, 64x32 each.
// All three tiles need fp32->bf16 conversion -> register double-buffering.
// ---------------------------------------------------------------------------
__global__ __launch_bounds__(256) void moe_gemm1_swiglu(
    const float* __restrict__ x, const float* __restrict__ w1,
    const float* __restrict__ w3, unsigned short* __restrict__ act)
{
  __shared__ __align__(16) unsigned short sA [128 * LPITCH];
  __shared__ __align__(16) unsigned short sB1[128 * LPITCH];
  __shared__ __align__(16) unsigned short sB3[128 * LPITCH];

  const int t    = threadIdx.x;
  const int lane = t & 31;
  const int wave = t >> 5;
  const int wm   = wave & 1;        // 0..1 along M
  const int wn   = wave >> 1;       // 0..3 along N
  const int e    = blockIdx.z;
  const int m0   = blockIdx.y * 128;
  const int n0   = blockIdx.x * 128;
  const int b    = m0 >> 9;         // 128 | 512 so a tile never straddles b
  const int s0   = m0 & 511;

  const float* xbase  = x  + (((size_t)b * E_ + e) * S_ + s0) * H_;
  const float* w1base = w1 + (size_t)e * I_ * H_ + (size_t)n0 * H_;
  const float* w3base = w3 + (size_t)e * I_ * H_ + (size_t)n0 * H_;

  const int lrow = t >> 3;          // 0..31 (4 row passes of 32)
  const int lcol = (t & 7) * 4;     // K offset (floats == ushorts)

  const v8f zacc = {0.f, 0.f, 0.f, 0.f, 0.f, 0.f, 0.f, 0.f};
  v8f accG[4][2], accU[4][2];
#pragma unroll
  for (int i = 0; i < 4; ++i)
#pragma unroll
    for (int j = 0; j < 2; ++j) { accG[i][j] = zacc; accU[i][j] = zacc; }

  uint2 ra[4], rb1[4], rb3[4];
  auto load_tiles = [&](int k0) {
#pragma unroll
    for (int p = 0; p < 4; ++p) {
      int r = lrow + p * 32;
      ra [p] = f2bf4(*(const float4*)(xbase  + (size_t)r * H_ + k0 + lcol));
      rb1[p] = f2bf4(*(const float4*)(w1base + (size_t)r * H_ + k0 + lcol));
      rb3[p] = f2bf4(*(const float4*)(w3base + (size_t)r * H_ + k0 + lcol));
    }
  };
  auto store_tiles = [&]() {
#pragma unroll
    for (int p = 0; p < 4; ++p) {
      int r = lrow + p * 32;
      *(uint2*)&sA [r * LPITCH + lcol] = ra [p];
      *(uint2*)&sB1[r * LPITCH + lcol] = rb1[p];
      *(uint2*)&sB3[r * LPITCH + lcol] = rb3[p];
    }
  };

  load_tiles(0);
  const int KT = H_ / 32;           // 64
  for (int kt = 0; kt < KT; ++kt) {
    __syncthreads();
    store_tiles();
    __syncthreads();
    if (kt + 1 < KT) load_tiles((kt + 1) * 32);   // overlap with WMMA below

    v16bf af[4];
#pragma unroll
    for (int im = 0; im < 4; ++im)
      af[im] = frag_a(sA, wm * 64 + im * 16 + (lane & 15), lane);

#pragma unroll
    for (int in = 0; in < 2; ++in) {
      v16bf b1f = frag_b(sB1, wn * 32 + in * 16 + (lane & 15), lane);
      v16bf b3f = frag_b(sB3, wn * 32 + in * 16 + (lane & 15), lane);
#pragma unroll
      for (int im = 0; im < 4; ++im) {
        accG[im][in] = __builtin_amdgcn_wmma_f32_16x16x32_bf16(
            false, af[im], false, b1f, (short)0, accG[im][in], false, false);
        accU[im][in] = __builtin_amdgcn_wmma_f32_16x16x32_bf16(
            false, af[im], false, b3f, (short)0, accU[im][in], false, false);
      }
    }
  }

  // SwiGLU epilogue -> bf16 act[e][m][i]
  unsigned short* actb = act + (size_t)e * MPE * I_;
#pragma unroll
  for (int im = 0; im < 4; ++im) {
#pragma unroll
    for (int in = 0; in < 2; ++in) {
      v8f g = accG[im][in];
      v8f u = accU[im][in];
#pragma unroll
      for (int r = 0; r < 8; ++r) {
        float gv = g[r];
        float sg = __builtin_amdgcn_rcpf(1.f + __expf(-gv));  // sigmoid via v_rcp_f32
        float a  = gv * sg * u[r];                            // silu(g) * up
        int m = m0 + wm * 64 + im * 16 + r + ((lane & 16) ? 8 : 0);
        int i = n0 + wn * 32 + in * 16 + (lane & 15);
        actb[(size_t)m * I_ + i] = f2bf(a);
      }
    }
  }
}

// ---------------------------------------------------------------------------
// Kernel 2: out[b,e,s,h] = act[e,m,:] @ w2[e,:,h]   (K = I = 1408)
// A-tile (bf16 act) is copied with GLOBAL_LOAD_ASYNC_TO_LDS_B128 into a
// double-buffered LDS tile (no conversion, no VGPR round-trip, ASYNCcnt sync).
// w2 tile is [K][N] fp32 -> bf16 via VGPRs, transposed while staging to LDS.
// ---------------------------------------------------------------------------
__global__ __launch_bounds__(256) void moe_gemm2(
    const unsigned short* __restrict__ act, const float* __restrict__ w2,
    float* __restrict__ out)
{
  __shared__ __align__(16) unsigned short sA2[2][128 * LPITCH];  // async-filled
  __shared__ __align__(16) unsigned short sB [128 * LPITCH];

  const int t    = threadIdx.x;
  const int lane = t & 31;
  const int wave = t >> 5;
  const int wm   = wave & 1;
  const int wn   = wave >> 1;
  const int e    = blockIdx.z;
  const int m0   = blockIdx.y * 128;
  const int n0   = blockIdx.x * 128;   // along H

  const unsigned short* abase = act + ((size_t)e * MPE + m0) * I_;
  const float* w2base = w2 + (size_t)e * I_ * H_;

  // LDS byte addresses (LDS aperture: generic addr[31:0] == LDS offset)
  const unsigned ldsA0 = (unsigned)(uintptr_t)&sA2[0][0];
  const unsigned ldsA1 = (unsigned)(uintptr_t)&sA2[1][0];

  const v8f zacc = {0.f, 0.f, 0.f, 0.f, 0.f, 0.f, 0.f, 0.f};
  v8f acc[4][2];
#pragma unroll
  for (int i = 0; i < 4; ++i)
#pragma unroll
    for (int j = 0; j < 2; ++j) acc[i][j] = zacc;

  // A tile: 2 x 16B async copies per thread, direct to LDS
  auto issue_async_a = [&](int k0, int buf) {
    unsigned base = buf ? ldsA1 : ldsA0;
#pragma unroll
    for (int p = 0; p < 2; ++p) {
      int idx = t + p * 256;
      int r = idx >> 2;                 // 0..127
      int c = (idx & 3) * 8;            // ushort offset within 32-wide K
      async_copy_b128(base + (unsigned)((r * LPITCH + c) * 2),
                      abase + (size_t)r * I_ + k0 + c);
    }
  };

  // w2 tile: fp32 -> bf16 in VGPRs, transposed at LDS store
  ushort4 rB[4];
  auto load_w2 = [&](int k0) {
#pragma unroll
    for (int p = 0; p < 4; ++p) {
      int idx = t + p * 256;
      int kk = idx >> 5;                // 0..31
      int nb = (idx & 31) * 4;          // 0..124
      float4 v = *(const float4*)(w2base + (size_t)(k0 + kk) * H_ + n0 + nb);
      rB[p].x = f2bf(v.x); rB[p].y = f2bf(v.y);
      rB[p].z = f2bf(v.z); rB[p].w = f2bf(v.w);
    }
  };
  auto store_w2 = [&]() {
#pragma unroll
    for (int p = 0; p < 4; ++p) {
      int idx = t + p * 256;
      int kk = idx >> 5;
      int nb = (idx & 31) * 4;
      sB[(nb + 0) * LPITCH + kk] = rB[p].x;   // transpose: [K][N] -> [N][K]
      sB[(nb + 1) * LPITCH + kk] = rB[p].y;
      sB[(nb + 2) * LPITCH + kk] = rB[p].z;
      sB[(nb + 3) * LPITCH + kk] = rB[p].w;
    }
  };

  issue_async_a(0, 0);
  load_w2(0);
  const int KT = I_ / 32;               // 44
  for (int kt = 0; kt < KT; ++kt) {
    __syncthreads();                    // prev iter done reading sB / sA2[cur]
    store_w2();
    if (kt + 1 < KT) {
      issue_async_a((kt + 1) * 32, (kt + 1) & 1);  // fill other buffer
      load_w2((kt + 1) * 32);
      WAIT_ASYNCCNT(2);                 // allow only the 2 just-issued copies
    } else {
      WAIT_ASYNCCNT(0);
    }
    __syncthreads();                    // sB stores + everyone's async tile done

    const unsigned short* sAcur = &sA2[kt & 1][0];
    v16bf af[4];
#pragma unroll
    for (int im = 0; im < 4; ++im)
      af[im] = frag_a(sAcur, wm * 64 + im * 16 + (lane & 15), lane);
#pragma unroll
    for (int in = 0; in < 2; ++in) {
      v16bf bf = frag_b(sB, wn * 32 + in * 16 + (lane & 15), lane);
#pragma unroll
      for (int im = 0; im < 4; ++im) {
        acc[im][in] = __builtin_amdgcn_wmma_f32_16x16x32_bf16(
            false, af[im], false, bf, (short)0, acc[im][in], false, false);
      }
    }
  }

  // fp32 epilogue -> out[b,e,s,h]; non-temporal (written once, never re-read)
#pragma unroll
  for (int im = 0; im < 4; ++im) {
#pragma unroll
    for (int in = 0; in < 2; ++in) {
      v8f a = acc[im][in];
#pragma unroll
      for (int r = 0; r < 8; ++r) {
        int m  = m0 + wm * 64 + im * 16 + r + ((lane & 16) ? 8 : 0);
        int bb = m >> 9;
        int s  = m & 511;
        int h  = n0 + wn * 32 + in * 16 + (lane & 15);
        __builtin_nontemporal_store(
            a[r], &out[(((size_t)bb * E_ + e) * S_ + s) * H_ + h]);
      }
    }
  }
}

extern "C" void kernel_launch(void* const* d_in, const int* in_sizes, int n_in,
                              void* d_out, int out_size, void* d_ws, size_t ws_size,
                              hipStream_t stream) {
  // setup_inputs() dict order: hidden_states, w1, w3, w2
  const float* x  = (const float*)d_in[0];
  const float* w1 = (const float*)d_in[1];
  const float* w3 = (const float*)d_in[2];
  const float* w2 = (const float*)d_in[3];
  float* out = (float*)d_out;

  // workspace: bf16 act [E][MPE][I] = 16*1024*1408*2 B ~= 46.1 MB
  unsigned short* act = (unsigned short*)d_ws;

  dim3 blk(256);
  dim3 g1(I_ / 128, MPE / 128, E_);   // 11 x 8 x 16
  moe_gemm1_swiglu<<<g1, blk, 0, stream>>>(x, w1, w3, act);

  dim3 g2(H_ / 128, MPE / 128, E_);   // 16 x 8 x 16
  moe_gemm2<<<g2, blk, 0, stream>>>(act, w2, out);
}
